// DeformConv_1726576855658
// MI455X (gfx1250) — matrix operand order repaired and checked by reading, not compiled, not run on hardware
//
#include <hip/hip_runtime.h>
#include <hip/hip_bf16.h>

typedef __attribute__((ext_vector_type(16))) __bf16 v16bf;
typedef __attribute__((ext_vector_type(8)))  __bf16 v8bf;
typedef __attribute__((ext_vector_type(8)))  float  v8f;

// Problem constants (match reference): B=4, C=256, H=W=64, Co=256, 3x3, pad=1
#define NB   4
#define NC   256
#define NH   64
#define NW   64
#define NCO  256
#define NPIX (NB * NH * NW)        // 16384
#define KDIM (NC * 9)              // 2304
#define KSTEPS (KDIM / 32)         // 72
#define WA_ELEMS (KSTEPS * 16 * 32 * 16)   // 589824 bf16 elems (A-frag swizzled)
#define XT_BYTES ((size_t)NB * NH * NW * NC * 4)  // 16.78 MB NHWC fp32

// ---------------------------------------------------------------------------
// Kernel 1: NCHW fp32 -> NHWC fp32 tiled transpose (one-time, L2-resident).
// grid (S/32=128, C/32=8, B=4), 256 threads.
// ---------------------------------------------------------------------------
__global__ __launch_bounds__(256) void prep_x_nhwc(const float* __restrict__ in,
                                                   float* __restrict__ outT) {
  __shared__ float tile[32][33];
  const int b    = blockIdx.z;
  const int cBlk = blockIdx.y << 5;
  const int sBlk = blockIdx.x << 5;
  const int tx = threadIdx.x & 31;
  const int ty = threadIdx.x >> 5;              // 0..7
#pragma unroll
  for (int i = 0; i < 32; i += 8)
    tile[ty + i][tx] = in[(((b * NC) + cBlk + ty + i) << 12) + sBlk + tx];
  __syncthreads();
#pragma unroll
  for (int i = 0; i < 32; i += 8)
    outT[(((b << 12) + sBlk + ty + i) << 8) + cBlk + tx] = tile[tx][ty + i];
}

// ---------------------------------------------------------------------------
// Kernel 2: weight fp32 [Co,C,3,3] -> A-fragment-swizzled bf16 hi/lo planes.
// Column index kk = tap*256 + c. A-frag layout per ISA 7.12.2 (16x32 bf16):
//   lane<16 : M=lane,  K = {0..7, 16..23};  lane>=16: M=lane-16, K = {8..15, 24..31}
//   within a lane, element j (j=0..15): vgpr=j/2, half=j%2,
//   K = (j/8)*16 + (lane/16)*8 + ((j/2)%4)*2 + (j%2)
// Storage: wA[((step*16 + mtile)*32 + lane)*16 + j]
// ---------------------------------------------------------------------------
__global__ __launch_bounds__(256) void prep_w_swizzle(const float* __restrict__ w,
                                                      __bf16* __restrict__ hi,
                                                      __bf16* __restrict__ lo) {
  const int t = blockIdx.x * 256 + threadIdx.x;      // < 589824
  const int j     = t & 15;
  const int lane  = (t >> 4) & 31;
  const int mtile = (t >> 9) & 15;
  const int step  = t >> 13;
  const int co       = (mtile << 4) + (lane & 15);
  const int laneHalf = lane >> 4;
  const int vg = j >> 1, half = j & 1, grp = vg >> 2;
  const int Klocal = (grp << 4) + (laneHalf << 3) + ((vg & 3) << 1) + half;
  const int kk  = (step << 5) + Klocal;
  const int tap = kk >> 8;
  const int c   = kk & 255;
  const float v = w[(co * NC + c) * 9 + tap];
  const __bf16 h = (__bf16)v;
  hi[t] = h;
  lo[t] = (__bf16)(v - (float)h);
}

// ---------------------------------------------------------------------------
// Kernel 3: fused bilinear-sample + implicit GEMM via WMMA (bf16 hi/lo x3),
// double-buffered B tile in LDS (gather for step s+1 overlaps WMMA of step s).
// Block: 256 threads (8 wave32). Output tile: 128 Co x 64 pixels.
// grid.x = (NPIX/64) * 2 = 512.
// ---------------------------------------------------------------------------
__global__ __launch_bounds__(256) void deform_wmma(const float* __restrict__ xT,
                                                   const __bf16* __restrict__ wAhi,
                                                   const __bf16* __restrict__ wAlo,
                                                   const float* __restrict__ offset,
                                                   float* __restrict__ out) {
  __shared__ int4   sAddr[576];        // [tap][p]: 4 clamped corner offsets (elem)
  __shared__ float4 sWt[576];          // [tap][p]: 4 bilinear weights (0 if invalid)
  __shared__ v16bf  sBhi[2][4][32];    // double-buffered B fragments (hi plane)
  __shared__ v16bf  sBlo[2][4][32];    // double-buffered B fragments (lo plane)

  const int tid     = threadIdx.x;
  const int lane    = tid & 31;
  const int wave    = tid >> 5;
  const int pixTile = blockIdx.x >> 1;
  const int mblock  = blockIdx.x & 1;

  // ---- per-block sampling precompute: 9 taps x 64 pixels ----
  for (int i = tid; i < 576; i += 256) {
    const int tap = i >> 6;
    const int p   = i & 63;
    const int pg  = (pixTile << 6) + p;
    const int b   = pg >> 12;
    const int rem = pg & 4095;
    const int y   = rem >> 6;
    const int xq  = rem & 63;
    const int ky = tap / 3, kx = tap % 3;
    const float dy = offset[(((b * 18) + 2 * tap)     << 12) + rem];
    const float dx = offset[(((b * 18) + 2 * tap + 1) << 12) + rem];
    const float sy = (float)(y - 1 + ky) + dy;
    const float sx = (float)(xq - 1 + kx) + dx;
    const float fy0 = floorf(sy), fx0 = floorf(sx);
    const int y0 = (int)fy0, x0 = (int)fx0;
    const int y1 = y0 + 1,   x1 = x0 + 1;
    const float wy1 = sy - fy0, wx1 = sx - fx0;
    const float wy0 = 1.0f - wy1, wx0 = 1.0f - wx1;
    const bool vy0 = (y0 >= 0) & (y0 < NH), vy1 = (y1 >= 0) & (y1 < NH);
    const bool vx0 = (x0 >= 0) & (x0 < NW), vx1 = (x1 >= 0) & (x1 < NW);
    const int yc0 = min(max(y0, 0), NH - 1), yc1 = min(max(y1, 0), NH - 1);
    const int xc0 = min(max(x0, 0), NW - 1), xc1 = min(max(x1, 0), NW - 1);
    const int base = b << 12;
    sAddr[i] = make_int4(((base + (yc0 << 6) + xc0) << 8),
                         ((base + (yc0 << 6) + xc1) << 8),
                         ((base + (yc1 << 6) + xc0) << 8),
                         ((base + (yc1 << 6) + xc1) << 8));
    sWt[i] = make_float4(wy0 * wx0 * (float)(vy0 && vx0),
                         wy0 * wx1 * (float)(vy0 && vx1),
                         wy1 * wx0 * (float)(vy1 && vx0),
                         wy1 * wx1 * (float)(vy1 && vx1));
  }

  v8f acc[4];
#pragma unroll
  for (int ns = 0; ns < 4; ++ns) acc[ns] = (v8f){};

  // column-builder thread mapping: 4 threads per pixel, 8 channels each
  const int p_local   = tid >> 2;
  const int cg        = tid & 3;
  const int writeLane = (p_local & 15) + ((cg & 1) << 4);   // lane = (N&15)+16*((K/8)&1)
  const int nsub_w    = p_local >> 4;
  const int jhalf     = cg >> 1;                             // 8-elem half of v16bf
  const int mtileG    = (mblock << 3) + wave;

  // Build B tile for `step` into buffer `buf` (fp32 gather -> bf16 hi/lo split).
  auto buildB = [&](int step, int buf) {
    const int tap = step >> 3;
    const int c0  = ((step & 7) << 5) + (cg << 3);
    const int rec = (tap << 6) + p_local;
    const int4   ad = sAddr[rec];
    const float4 wv = sWt[rec];
    const float4* q0 = (const float4*)(xT + ad.x + c0);
    const float4* q1 = (const float4*)(xT + ad.y + c0);
    const float4* q2 = (const float4*)(xT + ad.z + c0);
    const float4* q3 = (const float4*)(xT + ad.w + c0);
    const float4 a0 = q0[0], a1 = q0[1];
    const float4 b0 = q1[0], b1 = q1[1];
    const float4 c0v = q2[0], c1 = q2[1];
    const float4 d0 = q3[0], d1 = q3[1];
    float col[8];
    col[0] = wv.x * a0.x + wv.y * b0.x + wv.z * c0v.x + wv.w * d0.x;
    col[1] = wv.x * a0.y + wv.y * b0.y + wv.z * c0v.y + wv.w * d0.y;
    col[2] = wv.x * a0.z + wv.y * b0.z + wv.z * c0v.z + wv.w * d0.z;
    col[3] = wv.x * a0.w + wv.y * b0.w + wv.z * c0v.w + wv.w * d0.w;
    col[4] = wv.x * a1.x + wv.y * b1.x + wv.z * c1.x + wv.w * d1.x;
    col[5] = wv.x * a1.y + wv.y * b1.y + wv.z * c1.y + wv.w * d1.y;
    col[6] = wv.x * a1.z + wv.y * b1.z + wv.z * c1.z + wv.w * d1.z;
    col[7] = wv.x * a1.w + wv.y * b1.w + wv.z * c1.w + wv.w * d1.w;
    v8bf chi, clo;
#pragma unroll
    for (int j = 0; j < 8; ++j) {
      const __bf16 h = (__bf16)col[j];
      chi[j] = h;
      clo[j] = (__bf16)(col[j] - (float)h);
    }
    ((v8bf*)&sBhi[buf][nsub_w][writeLane])[jhalf] = chi;
    ((v8bf*)&sBlo[buf][nsub_w][writeLane])[jhalf] = clo;
  };

  // prologue: fill buffer 0
  __syncthreads();          // sampling records visible
  buildB(0, 0);
  __syncthreads();

  for (int step = 0; step < KSTEPS; ++step) {
    const int cur = step & 1;
    // issue next step's gather early so its VMEM latency hides under the WMMAs
    if (step + 1 < KSTEPS) buildB(step + 1, cur ^ 1);

    // ---- A fragments: one 32B vector load per lane per plane ----
    const int aIdx = ((step * 16 + mtileG) << 5) + lane;
    const v16bf ahi = ((const v16bf*)wAhi)[aIdx];
    const v16bf alo = ((const v16bf*)wAlo)[aIdx];

    // ---- 4 N-subtiles x 3 WMMAs (hi*hi + hi*lo + lo*hi) ----
#pragma unroll
    for (int ns = 0; ns < 4; ++ns) {
      const v16bf bhi = sBhi[cur][ns][lane];
      const v16bf blo = sBlo[cur][ns][lane];
      acc[ns] = __builtin_amdgcn_wmma_f32_16x16x32_bf16(
          false, ahi, false, bhi, (short)0, acc[ns], false, false);
      acc[ns] = __builtin_amdgcn_wmma_f32_16x16x32_bf16(
          false, ahi, false, blo, (short)0, acc[ns], false, false);
      acc[ns] = __builtin_amdgcn_wmma_f32_16x16x32_bf16(
          false, alo, false, bhi, (short)0, acc[ns], false, false);
    }
    __syncthreads();  // next buffer fully written; current buffer fully consumed
  }

  // ---- store D: (M,N) -> vgpr r, lane = (N&15)+16*(M/8), M = r + 8*(lane/16) ----
  const int laneN  = lane & 15;
  const int laneHi = lane >> 4;
#pragma unroll
  for (int ns = 0; ns < 4; ++ns) {
    const int pg = (pixTile << 6) + (ns << 4) + laneN;
    const int b  = pg >> 12;
    const int s  = pg & 4095;
#pragma unroll
    for (int r = 0; r < 8; ++r) {
      const int co = (mblock << 7) + (wave << 4) + (laneHi << 3) + r;
      out[(((b << 8) + co) << 12) + s] = acc[ns][r];
    }
  }
}

// ---------------------------------------------------------------------------
extern "C" void kernel_launch(void* const* d_in, const int* in_sizes, int n_in,
                              void* d_out, int out_size, void* d_ws, size_t ws_size,
                              hipStream_t stream) {
  const float* x      = (const float*)d_in[0];   // [4,256,64,64]
  const float* offset = (const float*)d_in[1];   // [4,18,64,64]
  const float* weight = (const float*)d_in[2];   // [256,256,3,3]
  float* out = (float*)d_out;

  float*  xT   = (float*)d_ws;
  __bf16* wAhi = (__bf16*)((char*)d_ws + XT_BYTES);
  __bf16* wAlo = wAhi + WA_ELEMS;

  prep_x_nhwc<<<dim3(NH * NW / 32, NC / 32, NB), 256, 0, stream>>>(x, xT);
  prep_w_swizzle<<<WA_ELEMS / 256, 256, 0, stream>>>(weight, wAhi, wAlo);
  deform_wmma<<<(NPIX / 64) * 2, 256, 0, stream>>>(xT, wAhi, wAlo, offset, out);
}
